// PointTransformerLayer_7473243095306
// MI455X (gfx1250) — matrix operand order repaired and checked by reading, not compiled, not run on hardware
//
#include <hip/hip_runtime.h>
#include <stdint.h>

// Problem constants (match reference)
#define BATCH 4
#define CIN   64
#define COUT  64
#define CCH   3
#define NPTS  8192
#define KNN   16

typedef __attribute__((ext_vector_type(16))) __bf16 v16bf;
typedef __attribute__((ext_vector_type(8)))  float  v8f;
typedef __attribute__((ext_vector_type(4)))  unsigned int v4u;

union Frag {
  v16bf    bf;
  uint32_t u[8];
  v4u      u4[2];
};

// Hardware f32 -> bf16 (RNE) via native conversion
__device__ __forceinline__ uint16_t f2bf(float x) {
  __bf16 h = (__bf16)x;
  return __builtin_bit_cast(uint16_t, h);
}
// 16-bit A matrix (16x32): VGPR v, lane-half h -> first K of packed pair
__device__ __forceinline__ int a_pair_k(int v, int half) {
  int base = 2 * v;  // {0,2,...,14}, contiguous
  return base + (half ? 16 : 0);
}
// 16-bit B matrix (32x16): VGPR j, lane-half h -> first K of packed pair
__device__ __forceinline__ int b_pair_k(int j, int half) {
  return 2 * j + (half ? 16 : 0);
}
__device__ __forceinline__ v8f wmma_bf16(const Frag& a, const Frag& b, v8f c) {
  return __builtin_amdgcn_wmma_f32_16x16x32_bf16(false, a.bf, false, b.bf,
                                                 (short)0, c, false, false);
}
// Wave-local LDS ordering (per-wave private buffers, in-order DS per wave)
__device__ __forceinline__ void lds_fence() {
  asm volatile("s_wait_dscnt 0x0" ::: "memory");
}
// Async global->LDS 16-byte copy (per-lane), tracked by ASYNCcnt
__device__ __forceinline__ void async_ld_b128(uint32_t lds_off,
                                              const void* gsrc) {
  asm volatile("global_load_async_to_lds_b128 %0, %1, off"
               :: "v"(lds_off), "v"((uint64_t)(uintptr_t)gsrc)
               : "memory");
}
__device__ __forceinline__ void wait_async0() {
  asm volatile("s_wait_asynccnt 0x0" ::: "memory");
}

// ---------------------------------------------------------------------------
// K0a: features (B,C,N) f32 -> fbf (B,N,C) bf16 transpose + squared norms
// ---------------------------------------------------------------------------
__global__ __launch_bounds__(256) void k0_convert(
    const float* __restrict__ feat, uint16_t* __restrict__ fbf,
    float* __restrict__ sqn) {
  int t = blockIdx.x * blockDim.x + threadIdx.x;
  if (t >= BATCH * NPTS) return;
  int b = t / NPTS, n = t % NPTS;
  const float* src = feat + (size_t)b * CIN * NPTS + n;
  uint16_t* dst = fbf + (size_t)t * CIN;
  float s = 0.f;
#pragma unroll 8
  for (int c = 0; c < CIN; ++c) {
    float v = src[(size_t)c * NPTS];
    s += v * v;
    dst[c] = f2bf(v);
  }
  sqn[t] = s;
}

// ---------------------------------------------------------------------------
// K0b: one-time f32 -> bf16 conversion of the six 64x64 weight matrices.
// Order: [0]=phi [1]=psi [2]=alpha [3]=d2 [4]=g1 [5]=g2
// ---------------------------------------------------------------------------
__global__ __launch_bounds__(256) void k0_weights(
    const float* __restrict__ Wphi, const float* __restrict__ Wpsi,
    const float* __restrict__ Wal,  const float* __restrict__ Wd2,
    const float* __restrict__ Wg1,  const float* __restrict__ Wg2,
    uint16_t* __restrict__ wb) {
  int i = blockIdx.x * blockDim.x + threadIdx.x;
  if (i >= COUT * CIN) return;
  wb[0 * COUT * CIN + i] = f2bf(Wphi[i]);
  wb[1 * COUT * CIN + i] = f2bf(Wpsi[i]);
  wb[2 * COUT * CIN + i] = f2bf(Wal[i]);
  wb[3 * COUT * CIN + i] = f2bf(Wd2[i]);
  wb[4 * COUT * CIN + i] = f2bf(Wg1[i]);
  wb[5 * COUT * CIN + i] = f2bf(Wg2[i]);
}

// ---------------------------------------------------------------------------
// K1: phi/psi/alpha = W(64x64) * F(64xN) + b via bf16 WMMA, fp32 accumulate.
// Outputs stored TRANSPOSED as (B,N,C) so K3's KNN gathers become per-lane
// contiguous 32-byte runs. grid (N/16, B), block 128 (4 waves).
// ---------------------------------------------------------------------------
__global__ __launch_bounds__(128) void k1_qkv(
    const uint16_t* __restrict__ fbf, const uint16_t* __restrict__ wb,
    const float* __restrict__ bphi, const float* __restrict__ bpsi,
    const float* __restrict__ bal,
    float* __restrict__ phiT, float* __restrict__ psiT,
    float* __restrict__ alphaT) {
  const int wave = threadIdx.x >> 5;
  const int lane = threadIdx.x & 31;
  const int half = lane >> 4;
  const int lcol = lane & 15;
  const int b = blockIdx.y;
  const int nbase = blockIdx.x * 16;
  const int obase = wave * 16;

  // B fragments: F tile (32 in-ch x 16 cols), pairs contiguous in c in fbf
  Frag fb[2];
#pragma unroll
  for (int ks = 0; ks < 2; ++ks) {
    const uint16_t* src =
        fbf + (size_t)(b * NPTS + nbase + lcol) * CIN + ks * 32;
#pragma unroll
    for (int j = 0; j < 8; ++j)
      fb[ks].u[j] = *(const uint32_t*)(src + b_pair_k(j, half));
  }

  const uint16_t* wP = wb + 0 * COUT * CIN;
  const uint16_t* wS = wb + 1 * COUT * CIN;
  const uint16_t* wA = wb + 2 * COUT * CIN;

  v8f accP = {}, accS = {}, accA = {};
  const int o = obase + lcol;  // A-row = out channel
#pragma unroll
  for (int ks = 0; ks < 2; ++ks) {
    Frag faP, faS, faA;
#pragma unroll
    for (int v = 0; v < 8; ++v) {
      int kk = ks * 32 + a_pair_k(v, half);
      faP.u[v] = *(const uint32_t*)(wP + o * CIN + kk);
      faS.u[v] = *(const uint32_t*)(wS + o * CIN + kk);
      faA.u[v] = *(const uint32_t*)(wA + o * CIN + kk);
    }
    accP = wmma_bf16(faP, fb[ks], accP);
    accS = wmma_bf16(faS, fb[ks], accS);
    accA = wmma_bf16(faA, fb[ks], accA);
  }
  // Transposed store: lane writes 8 consecutive channels of point nbase+lcol
  size_t rowT = ((size_t)(b * NPTS) + nbase + lcol) * COUT + obase + half * 8;
#pragma unroll
  for (int r = 0; r < 8; ++r) {
    int oo = obase + half * 8 + r;
    phiT[rowT + r]   = accP[r] + bphi[oo];
    psiT[rowT + r]   = accS[r] + bpsi[oo];
    alphaT[rowT + r] = accA[r] + bal[oo];
  }
}

// ---------------------------------------------------------------------------
// K2: fused feature-space KNN. Candidate panels are staged block-wide in LDS
// via double-buffered async global->LDS DMA (ASYNCcnt), one b128 per thread
// per tile; all 4 waves share the panel. Gram tiles via WMMA. Both halves of
// each wave scan 8 candidates/query with private sorted top-16; merged at end.
// grid (N/64, B), block 128 (4 waves)
// ---------------------------------------------------------------------------
__global__ __launch_bounds__(128) void k2_knn(
    const uint16_t* __restrict__ fbf, const float* __restrict__ sqn,
    int* __restrict__ knn_idx) {
  __shared__ uint16_t panel[2][16 * CIN];  // double-buffered candidate tile
  __shared__ float dotT[4][16][16];
  __shared__ float sqT[4][16];
  __shared__ float mergeV[4][16][16];
  __shared__ int   mergeI[4][16][16];
  const int tid  = threadIdx.x;
  const int wave = tid >> 5;
  const int lane = tid & 31;
  const int half = lane >> 4;
  const int lcol = lane & 15;
  const int b = blockIdx.y;
  const int qbase = (blockIdx.x * 4 + wave) * 16;

  // A fragments: 16 queries x 64 channels (2 K-steps), loop invariant
  Frag fa[2];
#pragma unroll
  for (int ks = 0; ks < 2; ++ks) {
    const uint16_t* src =
        fbf + (size_t)(b * NPTS + qbase + lcol) * CIN + ks * 32;
#pragma unroll
    for (int v = 0; v < 8; ++v)
      fa[ks].u[v] = *(const uint32_t*)(src + a_pair_k(v, half));
  }
  const float sqq = sqn[b * NPTS + qbase + lcol];

  float vals[KNN];
  int   idxs[KNN];
#pragma unroll
  for (int s = 0; s < KNN; ++s) { vals[s] = -3.4e38f; idxs[s] = 0; }

  const int NT = NPTS / 16;
  const int jb = half * 8;  // candidate sub-range this half-wave scans

  // Prologue: stream tile 0 into panel[0] (2048B = one b128 per thread)
  async_ld_b128((uint32_t)(uintptr_t)&panel[0][tid * 8],
                fbf + (size_t)(b * NPTS) * CIN + tid * 8);

  int buf = 0;
  for (int t = 0; t < NT; ++t) {
    const int cbase = t * 16;
    wait_async0();     // own DMA into panel[buf] retired
    __syncthreads();   // => whole panel[buf] visible block-wide

    // B fragments from the shared panel (two ds_load_b128 per K-step)
    Frag fbr[2];
#pragma unroll
    for (int ks = 0; ks < 2; ++ks) {
      const uint16_t* pb =
          &panel[buf][lcol * CIN + ks * 32 + (half ? 16 : 0)];
      fbr[ks].u4[0] = *(const v4u*)(pb);
      fbr[ks].u4[1] = *(const v4u*)(pb + 8);
    }
    // Stream next tile into the other buffer while we compute
    if (t + 1 < NT)
      async_ld_b128((uint32_t)(uintptr_t)&panel[buf ^ 1][tid * 8],
                    fbf + (size_t)(b * NPTS + (t + 1) * 16) * CIN + tid * 8);

    v8f acc = {};
    acc = wmma_bf16(fa[0], fbr[0], acc);
    acc = wmma_bf16(fa[1], fbr[1], acc);

#pragma unroll
    for (int r = 0; r < 8; ++r)
      dotT[wave][r + (half ? 8 : 0)][lcol] = acc[r];
    if (lane < 16) sqT[wave][lane] = sqn[b * NPTS + cbase + lane];
    lds_fence();

#pragma unroll
    for (int j = 0; j < 8; ++j) {
      float nd = 2.f * dotT[wave][lcol][jb + j] - sqq - sqT[wave][jb + j];
      if (nd > vals[KNN - 1]) {
        float cv = nd; int ci = cbase + jb + j;
#pragma unroll
        for (int s = 0; s < KNN; ++s) {
          bool sw = cv > vals[s];
          float tv = vals[s]; int ti = idxs[s];
          vals[s] = sw ? cv : tv; idxs[s] = sw ? ci : ti;
          cv = sw ? tv : cv;      ci = sw ? ti : ci;
        }
      }
    }
    lds_fence();
    buf ^= 1;
  }

  // Merge the two half-wave top-16 lists per query.
  if (half) {
#pragma unroll
    for (int s = 0; s < KNN; ++s) {
      mergeV[wave][lcol][s] = vals[s];
      mergeI[wave][lcol][s] = idxs[s];
    }
  }
  lds_fence();
  if (!half) {
#pragma unroll 4
    for (int m = 0; m < KNN; ++m) {
      float cv = mergeV[wave][lcol][m];
      int   ci = mergeI[wave][lcol][m];
      if (cv > vals[KNN - 1]) {
#pragma unroll
        for (int s = 0; s < KNN; ++s) {
          bool sw = cv > vals[s];
          float tv = vals[s]; int ti = idxs[s];
          vals[s] = sw ? cv : tv; idxs[s] = sw ? ci : ti;
          cv = sw ? tv : cv;      ci = sw ? ti : ci;
        }
      }
    }
    int* dst = knn_idx + ((size_t)(b * NPTS) + qbase + lcol) * KNN;
#pragma unroll
    for (int s = 0; s < KNN; ++s) dst[s] = idxs[s];
  }
}

// ---------------------------------------------------------------------------
// K3: per-point aggregation. One wave per point: position-MLP (VALU, K=3),
// then d2/g1/g2 as 64x64 x 64x16 WMMA GEMMs (bf16 weights streamed from
// global, L2-resident). phi/psi/alpha are in transposed (B,N,C) layout so
// the KNN gathers are per-lane contiguous. Waves fully independent; all LDS
// is per-wave, synced with s_wait_dscnt.
// block 128 (4 waves), each wave processes 8 points.
// ---------------------------------------------------------------------------
#define WPB 4
#define PPW 8
__global__ __launch_bounds__(128) void k3_aggregate(
    const float* __restrict__ coords,
    const float* __restrict__ phiT, const float* __restrict__ psiT,
    const float* __restrict__ alphaT, const int* __restrict__ knn_idx,
    const uint16_t* __restrict__ wb,
    const float* __restrict__ Wd1, const float* __restrict__ bd1,
    const float* __restrict__ bd2, const float* __restrict__ bg1,
    const float* __restrict__ bg2, float* __restrict__ out) {
  __shared__ uint16_t xA[WPB][16 * COUT];  // activation panel (neighbor-major)
  __shared__ uint16_t xB[WPB][16 * COUT];
  __shared__ float    dls[WPB][COUT * 16]; // delta, fp32
  __shared__ float    gms[WPB][COUT * 16]; // gamma, fp32
  __shared__ int      nidx[WPB][16];

  const int wave = threadIdx.x >> 5;
  const int lane = threadIdx.x & 31;
  const int half = lane >> 4;
  const int lcol = lane & 15;

  const uint16_t* wD2 = wb + 3 * COUT * CIN;
  const uint16_t* wG1 = wb + 4 * COUT * CIN;
  const uint16_t* wG2 = wb + 5 * COUT * CIN;

  for (int it = 0; it < PPW; ++it) {
    const int p = (blockIdx.x * WPB + wave) * PPW + it;
    const int b = p / NPTS;
    const int n = p % NPTS;

    // Phase A: neighbor indices
    if (lane < 16)
      nidx[wave][lane] = knn_idx[((size_t)(b * NPTS) + n) * KNN + lane];
    lds_fence();

    // Phase B: h = relu(Wd1 * coords_space + bd1) -> xA (bf16, neighbor-major)
    {
      float cb0 = coords[((size_t)b * CCH + 0) * NPTS + n];
      float cb1 = coords[((size_t)b * CCH + 1) * NPTS + n];
      float cb2 = coords[((size_t)b * CCH + 2) * NPTS + n];
      int o0 = lane, o1 = lane + 32;
      float w00 = Wd1[o0 * 3], w01 = Wd1[o0 * 3 + 1], w02 = Wd1[o0 * 3 + 2];
      float w10 = Wd1[o1 * 3], w11 = Wd1[o1 * 3 + 1], w12 = Wd1[o1 * 3 + 2];
      float bb0 = bd1[o0], bb1 = bd1[o1];
      for (int j = 0; j < 16; ++j) {
        int id = nidx[wave][j];
        float c0 = cb0 - coords[((size_t)b * CCH + 0) * NPTS + id];
        float c1 = cb1 - coords[((size_t)b * CCH + 1) * NPTS + id];
        float c2 = cb2 - coords[((size_t)b * CCH + 2) * NPTS + id];
        float h0 = fmaf(w00, c0, fmaf(w01, c1, fmaf(w02, c2, bb0)));
        float h1 = fmaf(w10, c0, fmaf(w11, c1, fmaf(w12, c2, bb1)));
        xA[wave][j * COUT + o0] = f2bf(fmaxf(h0, 0.f));
        xA[wave][j * COUT + o1] = f2bf(fmaxf(h1, 0.f));
      }
    }
    lds_fence();

    const float* prow = phiT + ((size_t)(b * NPTS) + n) * COUT;

    // Phase C: delta = Wd2*h + bd2 ; s = (phi - psi[idx]) + delta -> xB
    {
      int id = -1;
      const float* srow = nullptr;
#pragma unroll
      for (int mt = 0; mt < 4; ++mt) {
        v8f acc = {};
#pragma unroll
        for (int ks = 0; ks < 2; ++ks) {
          Frag faW, fbX;
          int o = mt * 16 + lcol;
#pragma unroll
          for (int v = 0; v < 8; ++v)
            faW.u[v] = *(const uint32_t*)(wD2 + o * CIN + ks * 32 + a_pair_k(v, half));
#pragma unroll
          for (int j = 0; j < 8; ++j)
            fbX.u[j] = *(const uint32_t*)(xA[wave] + lcol * COUT + ks * 32 + b_pair_k(j, half));
          acc = wmma_bf16(faW, fbX, acc);
        }
        if (id < 0) {
          id = nidx[wave][lcol];
          srow = psiT + ((size_t)(b * NPTS) + id) * COUT;
        }
#pragma unroll
        for (int r = 0; r < 8; ++r) {
          int o = mt * 16 + half * 8 + r;
          float dd = acc[r] + bd2[o];
          dls[wave][o * 16 + lcol] = dd;
          float fs = prow[o] - srow[o];
          xB[wave][lcol * COUT + o] = f2bf(fs + dd);
        }
      }
    }
    lds_fence();

    // Phase D: t = relu(Wg1*s + bg1) -> xA
#pragma unroll
    for (int mt = 0; mt < 4; ++mt) {
      v8f acc = {};
#pragma unroll
      for (int ks = 0; ks < 2; ++ks) {
        Frag faW, fbX;
        int o = mt * 16 + lcol;
#pragma unroll
        for (int v = 0; v < 8; ++v)
          faW.u[v] = *(const uint32_t*)(wG1 + o * CIN + ks * 32 + a_pair_k(v, half));
#pragma unroll
        for (int j = 0; j < 8; ++j)
          fbX.u[j] = *(const uint32_t*)(xB[wave] + lcol * COUT + ks * 32 + b_pair_k(j, half));
        acc = wmma_bf16(faW, fbX, acc);
      }
#pragma unroll
      for (int r = 0; r < 8; ++r) {
        int o = mt * 16 + half * 8 + r;
        xA[wave][lcol * COUT + o] = f2bf(fmaxf(acc[r] + bg1[o], 0.f));
      }
    }
    lds_fence();

    // Phase E: gamma = Wg2*t + bg2 -> gms
#pragma unroll
    for (int mt = 0; mt < 4; ++mt) {
      v8f acc = {};
#pragma unroll
      for (int ks = 0; ks < 2; ++ks) {
        Frag faW, fbX;
        int o = mt * 16 + lcol;
#pragma unroll
        for (int v = 0; v < 8; ++v)
          faW.u[v] = *(const uint32_t*)(wG2 + o * CIN + ks * 32 + a_pair_k(v, half));
#pragma unroll
        for (int j = 0; j < 8; ++j)
          fbX.u[j] = *(const uint32_t*)(xA[wave] + lcol * COUT + ks * 32 + b_pair_k(j, half));
        acc = wmma_bf16(faW, fbX, acc);
      }
#pragma unroll
      for (int r = 0; r < 8; ++r) {
        int o = mt * 16 + half * 8 + r;
        gms[wave][o * 16 + lcol] = acc[r] + bg2[o];
      }
    }
    lds_fence();

    // Phase F: softmax over k, out = sum_k rho * (alpha + delta)
    const float* arow = alphaT + ((size_t)(b * NPTS) + n) * COUT;
#pragma unroll
    for (int h = 0; h < 2; ++h) {
      int o = lane + 32 * h;
      float mx = -3.4e38f;
#pragma unroll
      for (int j = 0; j < 16; ++j) mx = fmaxf(mx, gms[wave][o * 16 + j]);
      float ex[16];
      float sum = 0.f;
#pragma unroll
      for (int j = 0; j < 16; ++j) {
        ex[j] = __expf(gms[wave][o * 16 + j] - mx);
        sum += ex[j];
      }
      float inv = 1.f / sum;
      float al = arow[o];
      float acc = 0.f;
#pragma unroll
      for (int j = 0; j < 16; ++j)
        acc += ex[j] * inv * (al + dls[wave][o * 16 + j]);
      out[((size_t)b * COUT + o) * NPTS + n] = acc;
    }
    lds_fence();  // protect nidx/xA WAR before next point
  }
}

// ---------------------------------------------------------------------------
extern "C" void kernel_launch(void* const* d_in, const int* in_sizes, int n_in,
                              void* d_out, int out_size, void* d_ws,
                              size_t ws_size, hipStream_t stream) {
  (void)in_sizes; (void)n_in; (void)out_size; (void)ws_size;
  const float* feat  = (const float*)d_in[0];
  const float* coords= (const float*)d_in[1];
  const float* Wphi  = (const float*)d_in[2];  const float* bphi = (const float*)d_in[3];
  const float* Wpsi  = (const float*)d_in[4];  const float* bpsi = (const float*)d_in[5];
  const float* Wal   = (const float*)d_in[6];  const float* bal  = (const float*)d_in[7];
  const float* Wg1   = (const float*)d_in[8];  const float* bg1  = (const float*)d_in[9];
  const float* Wg2   = (const float*)d_in[10]; const float* bg2  = (const float*)d_in[11];
  const float* Wd1   = (const float*)d_in[12]; const float* bd1  = (const float*)d_in[13];
  const float* Wd2   = (const float*)d_in[14]; const float* bd2  = (const float*)d_in[15];

  size_t off = 0;
  auto take = [&](size_t bytes) -> void* {
    off = (off + 255) & ~(size_t)255;
    void* p = (char*)d_ws + off;
    off += bytes;
    return p;
  };
  uint16_t* fbf  = (uint16_t*)take((size_t)BATCH * NPTS * CIN * sizeof(uint16_t));
  float* sqn     = (float*)take((size_t)BATCH * NPTS * sizeof(float));
  float* phiT    = (float*)take((size_t)BATCH * NPTS * COUT * sizeof(float));
  float* psiT    = (float*)take((size_t)BATCH * NPTS * COUT * sizeof(float));
  float* alphaT  = (float*)take((size_t)BATCH * NPTS * COUT * sizeof(float));
  int* knn       = (int*)take((size_t)BATCH * NPTS * KNN * sizeof(int));
  uint16_t* wbf  = (uint16_t*)take((size_t)6 * COUT * CIN * sizeof(uint16_t));

  k0_convert<<<(BATCH * NPTS + 255) / 256, 256, 0, stream>>>(feat, fbf, sqn);
  k0_weights<<<(COUT * CIN + 255) / 256, 256, 0, stream>>>(
      Wphi, Wpsi, Wal, Wd2, Wg1, Wg2, wbf);

  dim3 g1(NPTS / 16, BATCH);
  k1_qkv<<<g1, 128, 0, stream>>>(fbf, wbf, bphi, bpsi, bal, phiT, psiT, alphaT);

  dim3 g2(NPTS / 16 / 4, BATCH);
  k2_knn<<<g2, 128, 0, stream>>>(fbf, sqn, knn);

  int blocks3 = (BATCH * NPTS) / (WPB * PPW);
  k3_aggregate<<<blocks3, 32 * WPB, 0, stream>>>(
      coords, phiT, psiT, alphaT, knn, wbf, Wd1, bd1, bd2, bg1, bg2,
      (float*)d_out);
}
